// MS_SSA_Conv_80178449482317
// MI455X (gfx1250) — matrix operand order repaired and checked by reading, not compiled, not run on hardware
//
#include <hip/hip_runtime.h>
#include <hip/hip_bf16.h>

typedef __attribute__((ext_vector_type(16))) _Float16 v16h;
typedef __attribute__((ext_vector_type(8)))  _Float16 v8h;
typedef __attribute__((ext_vector_type(8)))  float    v8f;
typedef __attribute__((ext_vector_type(4)))  int      v4i;

#define TT 4
#define BB 32
#define CC 256
#define HW 256
// M = TT*BB*HW = 32768 pixel-rows, K = N = 256

#if defined(__gfx1250__) && \
    __has_builtin(__builtin_amdgcn_global_load_async_to_lds_b128) && \
    __has_builtin(__builtin_amdgcn_s_wait_asynccnt)
#define USE_ASYNC_LDS 1
typedef __attribute__((address_space(1))) v4i gv4i;   // global int4
typedef __attribute__((address_space(3))) v4i lv4i;   // LDS int4
#else
#define USE_ASYNC_LDS 0
#endif

// ---------------------------------------------------------------------------
// K0: convert the three 256x256 f32 weight matrices to f16 (row-major [o,c])
// ---------------------------------------------------------------------------
__global__ __launch_bounds__(256) void k_wconv(const float* __restrict__ Wq,
                                               const float* __restrict__ Wk,
                                               const float* __restrict__ Wp,
                                               _Float16* __restrict__ Wh) {
  int i = blockIdx.x * 256 + threadIdx.x;           // 65536 threads
  Wh[i]            = (_Float16)Wq[i];
  Wh[65536  + i]   = (_Float16)Wk[i];
  Wh[131072 + i]   = (_Float16)Wp[i];
}

// ---------------------------------------------------------------------------
// K1: shortcut LIF on x -> spikes A[m,c] (f16, pixel-major), fused with
//     v-branch BN + LIF -> v output (binary f32) written in [T,B,h,N,Ch]
// ---------------------------------------------------------------------------
__global__ __launch_bounds__(256) void k_lif_x(const float* __restrict__ x,
                                               _Float16* __restrict__ A,
                                               float* __restrict__ vout,
                                               const float* __restrict__ vg,
                                               const float* __restrict__ vb,
                                               const float* __restrict__ vm,
                                               const float* __restrict__ vv) {
  int idx = blockIdx.x * 256 + threadIdx.x;         // B*HW*C = 2,097,152
  int c   = idx & 255;
  int hw  = (idx >> 8) & 255;
  int b   = idx >> 16;
  float inv = vg[c] * rsqrtf(vv[c] + 1e-5f);
  float add = vb[c] - vm[c] * inv;
  int h = c >> 5, ch = c & 31;
  float v1 = 0.f, v2 = 0.f;
#pragma unroll
  for (int t = 0; t < TT; ++t) {
    size_t tb = (size_t)t * BB + b;
    float xv = x[(tb * CC + c) * HW + hw];
    v1 = 0.5f * (v1 + xv);                          // v += (x - v)/tau, tau=2
    float s = (v1 >= 1.f) ? 1.f : 0.f;
    v1 *= (1.f - s);                                // hard reset
    A[(tb * HW + hw) * CC + c] = (_Float16)s;
    float y = s * inv + add;                        // bn_v(xs)
    v2 = 0.5f * (v2 + y);
    float s2 = (v2 >= 1.f) ? 1.f : 0.f;
    v2 *= (1.f - s2);
    vout[(tb * 8 + h) * 8192 + hw * 32 + ch] = s2;  // v output (binary)
  }
}

// ---------------------------------------------------------------------------
// K2: WMMA GEMM with LDS-staged weights.
//   MODE 0:  Yh = bn(A @ W^T)              -> f16 [M,256]
//   MODE 1:  Yf = bn(A @ W^T + bp)         -> f32 d_out in [T,B,C,H,W]
// 8 waves / block; each wave owns a 16-row A strip held in 64 VGPRs.
// Weights staged 64 output-columns at a time into LDS (padded stride 264
// halves -> conflict-free ds_load_b128 fragment reads), via async
// global->LDS loads when the gfx1250 builtins are available.
// ---------------------------------------------------------------------------
template <int MODE>
__global__ __launch_bounds__(256) void k_gemm(
    const _Float16* __restrict__ A, const _Float16* __restrict__ W,
    _Float16* __restrict__ Yh, float* __restrict__ Yf,
    const float* __restrict__ gam, const float* __restrict__ bet,
    const float* __restrict__ mu,  const float* __restrict__ va,
    const float* __restrict__ bp) {
  __shared__ _Float16 Blds[64 * 264];               // 33 KB, padded rows

  const int wave = threadIdx.x >> 5;
  const int lane = threadIdx.x & 31;
  const int r16  = lane & 15;
  const int hi   = lane >> 4;                       // lane half
  const int m0   = blockIdx.x * 128 + wave * 16;
  const int tb   = m0 >> 8;                         // (t*B+b) block (MODE 1)
  const int hwb  = m0 & 255;                        // pixel base     (MODE 1)

  // A fragments: lane holds row m0+r16; halves 0..7 = K kb+hi*8..+7,
  // halves 8..15 = K kb+16+hi*8..+7  (CDNA5 16-bit A 16x32 layout)
  v16h a[8];
  const _Float16* ap = A + (size_t)(m0 + r16) * CC;
#pragma unroll
  for (int kk = 0; kk < 8; ++kk) {
    union { v16h v; v8h h[2]; } u;
    u.h[0] = *(const v8h*)(ap + kk * 32 + hi * 8);
    u.h[1] = *(const v8h*)(ap + kk * 32 + hi * 8 + 16);
    a[kk] = u.v;
  }

  for (int grp = 0; grp < 4; ++grp) {               // 64 output cols / group
    __syncthreads();                                // LDS reuse fence
    // ---- stage W rows [grp*64, grp*64+64) into LDS ----
#pragma unroll
    for (int i = 0; i < 8; ++i) {                   // 2048 16B chunks / 256 thr
      int chunk = threadIdx.x + i * 256;
      int row = chunk >> 5;                         // 32 chunks per 512B row
      int c16 = chunk & 31;
      const _Float16* gsrc = W + (size_t)(grp * 64 + row) * CC + c16 * 8;
      _Float16* ldst = &Blds[row * 264 + c16 * 8];
#if USE_ASYNC_LDS
      __builtin_amdgcn_global_load_async_to_lds_b128(
          (gv4i*)(void*)const_cast<_Float16*>(gsrc),
          (lv4i*)(void*)ldst, 0, 0);
#else
      *(v8h*)ldst = *(const v8h*)gsrc;
#endif
    }
#if USE_ASYNC_LDS
    __builtin_amdgcn_s_wait_asynccnt(0);
#endif
    __syncthreads();

    // ---- 4 N-tiles of 16 columns from LDS ----
    for (int j = 0; j < 4; ++j) {
      int n = grp * 64 + j * 16 + r16;              // lane's output column
      const _Float16* bl = &Blds[(unsigned)(j * 16 + r16) * 264 + hi * 8];
      v8f acc = {};
#pragma unroll
      for (int kk = 0; kk < 8; ++kk) {
        union { v16h v; v8h h[2]; } u;              // B frag: K kb+hi*8..+7, kb+16+hi*8..+7
        u.h[0] = *(const v8h*)(bl + kk * 32);
        u.h[1] = *(const v8h*)(bl + kk * 32 + 16);
        acc = __builtin_amdgcn_wmma_f32_16x16x32_f16(
            false, a[kk], false, u.v, (short)0, acc, false, false);
      }
      float inv = gam[n] * rsqrtf(va[n] + 1e-5f);
      float add = bet[n] - mu[n] * inv;
      if (MODE == 0) {
#pragma unroll
        for (int r = 0; r < 8; ++r)                 // D: m = r + hi*8, col n
          Yh[(size_t)(m0 + r + hi * 8) * CC + n] =
              (_Float16)(acc[r] * inv + add);
      } else {
        float bias = bp[n];
#pragma unroll
        for (int r = 0; r < 8; ++r) {
          int hw = hwb + r + hi * 8;                // out[(tb*C + n)*HW + hw]
          Yf[((size_t)tb * CC + n) * HW + hw] = (acc[r] + bias) * inv + add;
        }
      }
    }
  }
}

// ---------------------------------------------------------------------------
// K3: in-place LIF over T on a [M,256] f16 buffer (q_sp / k_sp spikes)
// ---------------------------------------------------------------------------
__global__ __launch_bounds__(256) void k_lif_y(_Float16* __restrict__ Y) {
  int idx = blockIdx.x * 256 + threadIdx.x;         // B*HW*C = 2,097,152
  float v = 0.f;
#pragma unroll
  for (int t = 0; t < TT; ++t) {
    size_t o = (size_t)t * 2097152 + idx;           // t-stride = B*HW*C
    float y = (float)Y[o];
    v = 0.5f * (v + y);
    float s = (v >= 1.f) ? 1.f : 0.f;
    v *= (1.f - s);
    Y[o] = (_Float16)s;
  }
}

// ---------------------------------------------------------------------------
// K4a: kv_raw[t,b,c] = sum_hw k_sp * v_sp   (channel-aligned reduction)
// ---------------------------------------------------------------------------
__global__ __launch_bounds__(256) void k_kv_raw(const _Float16* __restrict__ Ksp,
                                                const float* __restrict__ Vout,
                                                float* __restrict__ kvr) {
  int idx = blockIdx.x * 256 + threadIdx.x;         // T*B*C = 32768
  int c = idx & 255;
  int b = (idx >> 8) & 31;
  int t = idx >> 13;
  int h = c >> 5, ch = c & 31;
  size_t tb = (size_t)t * BB + b;
  const _Float16* kp = Ksp + (tb * HW) * CC + c;
  const float*    vp = Vout + (tb * 8 + h) * 8192 + ch;
  float s = 0.f;
  for (int hw = 0; hw < HW; ++hw)
    s += (float)kp[(size_t)hw * CC] * vp[(size_t)hw * 32];
  kvr[idx] = s;
}

// K4b: talking-heads LIF over T, v_th = 0.5 -> binary kv_s (f16)
__global__ __launch_bounds__(256) void k_kv_lif(const float* __restrict__ kvr,
                                                _Float16* __restrict__ kvs) {
  int idx = blockIdx.x * 256 + threadIdx.x;         // B*C = 8192
  float v = 0.f;
#pragma unroll
  for (int t = 0; t < TT; ++t) {
    float y = kvr[(size_t)t * 8192 + idx];
    v = 0.5f * (v + y);
    float s = (v >= 0.5f) ? 1.f : 0.f;
    v *= (1.f - s);
    kvs[(size_t)t * 8192 + idx] = (_Float16)s;
  }
}

// K5: attn = q_sp * kv_s (broadcast over pixels), in place on Yq
__global__ __launch_bounds__(256) void k_attn(_Float16* __restrict__ Q,
                                              const _Float16* __restrict__ kvs) {
  int idx = blockIdx.x * 256 + threadIdx.x;         // T*B*HW*C = 8,388,608
  int c  = idx & 255;
  int tb = idx >> 16;                               // (t*B+b)
  Q[idx] = (_Float16)((float)Q[idx] * (float)kvs[(size_t)tb * 256 + c]);
}

// ---------------------------------------------------------------------------
extern "C" void kernel_launch(void* const* d_in, const int* in_sizes, int n_in,
                              void* d_out, int out_size, void* d_ws, size_t ws_size,
                              hipStream_t stream) {
  // setup_inputs() dict order
  const float* x   = (const float*)d_in[0];
  const float* Wq  = (const float*)d_in[1];
  const float* Wk  = (const float*)d_in[2];
  const float* Wp  = (const float*)d_in[3];
  const float* bp  = (const float*)d_in[4];
  const float* qg  = (const float*)d_in[5];
  const float* qb  = (const float*)d_in[6];
  const float* qm  = (const float*)d_in[7];
  const float* qv  = (const float*)d_in[8];
  const float* kg  = (const float*)d_in[9];
  const float* kb  = (const float*)d_in[10];
  const float* km  = (const float*)d_in[11];
  const float* kv  = (const float*)d_in[12];
  const float* vg  = (const float*)d_in[13];
  const float* vb  = (const float*)d_in[14];
  const float* vm  = (const float*)d_in[15];
  const float* vv  = (const float*)d_in[16];
  const float* pg  = (const float*)d_in[17];
  const float* pb  = (const float*)d_in[18];
  const float* pm  = (const float*)d_in[19];
  const float* pv  = (const float*)d_in[20];

  // workspace layout (~51 MB)
  char* ws = (char*)d_ws;
  _Float16* Wh  = (_Float16*)(ws);                                  // 3x64K halves
  _Float16* Axs = (_Float16*)(ws + 393216);                         // 16.78 MB
  _Float16* Yq  = (_Float16*)(ws + 393216 + 16777216);              // 16.78 MB
  _Float16* Yk  = (_Float16*)(ws + 393216 + 2 * 16777216);          // 16.78 MB
  float*    kvr = (float*)   (ws + 393216 + (size_t)3 * 16777216);  // 128 KB
  _Float16* kvs = (_Float16*)(ws + 393216 + (size_t)3 * 16777216 + 131072);

  float* out  = (float*)d_out;            // [T,B,C,H,W] = 8,388,608
  float* vout = out + 8388608;            // [T,B,h,N,Ch] = 8,388,608

  k_wconv<<<256, 256, 0, stream>>>(Wq, Wk, Wp, Wh);
  k_lif_x<<<8192, 256, 0, stream>>>(x, Axs, vout, vg, vb, vm, vv);
  k_gemm<0><<<256, 256, 0, stream>>>(Axs, Wh, Yq, nullptr,
                                     qg, qb, qm, qv, nullptr);
  k_gemm<0><<<256, 256, 0, stream>>>(Axs, Wh + 65536, Yk, nullptr,
                                     kg, kb, km, kv, nullptr);
  k_lif_y<<<8192, 256, 0, stream>>>(Yq);
  k_lif_y<<<8192, 256, 0, stream>>>(Yk);
  k_kv_raw<<<128, 256, 0, stream>>>(Yk, vout, kvr);
  k_kv_lif<<<32, 256, 0, stream>>>(kvr, kvs);
  k_attn<<<32768, 256, 0, stream>>>(Yq, kvs);
  k_gemm<1><<<256, 256, 0, stream>>>(Yq, Wh + 131072, nullptr, out,
                                     pg, pb, pm, pv, bp);
}
